// PVFC_51316269252905
// MI455X (gfx1250) — compile-verified, matrix operand order checked by reading, not run on hardware
//
#include <hip/hip_runtime.h>
#include <hip/hip_bf16.h>

// ---------------------------------------------------------------------------
// PV fully-connected layer for MI455X (gfx1250, wave32, WMMA).
//   v   = log0(x)            rowwise scale of x           (kernel: pvfc_log0, f16 out)
//   u   = v @ W^T            GEMM via v_wmma_f32_16x16x32_f16 (kernel: pvfc_gemm_wmma)
//   out = P(row)*u + Q(row)*b   (proj_tan0/exp0/pt/exp_map/gyro_add collapse to
//                                two per-row scalars)     (kernel: pvfc_epilogue)
// GEMM: branch-free 2-stage software pipeline over double-buffered LDS tiles
// filled by global_load_async_to_lds_b128 (ASYNCcnt), so the next tile's HBM
// fetch is always in flight under the current tile's WMMA work.
// Shapes fixed by the reference: N=16384, D_IN=D_OUT=2048 (K % 64 == 0).
// ---------------------------------------------------------------------------

typedef __attribute__((ext_vector_type(16))) _Float16 v16h;
typedef __attribute__((ext_vector_type(8)))  _Float16 v8h;
typedef __attribute__((ext_vector_type(4)))  _Float16 v4h;
typedef __attribute__((ext_vector_type(8)))  float    v8f;
typedef __attribute__((ext_vector_type(4)))  int      v4i;

#define EPSF  1e-6f
#define TINYF 1e-15f

#if __has_builtin(__builtin_amdgcn_global_load_async_to_lds_b128)
#define PVFC_USE_ASYNC 1
typedef __attribute__((address_space(1))) v4i* pvfc_g128_t;  // global int4*
typedef __attribute__((address_space(3))) v4i* pvfc_l128_t;  // LDS int4*
#endif

__device__ __forceinline__ void pvfc_async_wait0() {
#if __has_builtin(__builtin_amdgcn_s_wait_asynccnt)
  __builtin_amdgcn_s_wait_asynccnt(0);
#else
  asm volatile("s_wait_asynccnt 0" ::: "memory");
#endif
}

// Wait until at most 4 async ops outstanding: with 8 issued per thread
// (4 for tile i, then 4 for tile i+1) and in-order completion, this means
// tile i has fully landed in LDS while tile i+1 is still in flight.
__device__ __forceinline__ void pvfc_async_wait4() {
#if __has_builtin(__builtin_amdgcn_s_wait_asynccnt)
  __builtin_amdgcn_s_wait_asynccnt(4);
#else
  asm volatile("s_wait_asynccnt 4" ::: "memory");
#endif
}

// ---------------------------------------------------------------------------
// Kernel A: cast W (f32, row-major [D_OUT, D_IN]) -> f16
// ---------------------------------------------------------------------------
__global__ __launch_bounds__(256) void pvfc_cast_w(const float* __restrict__ W,
                                                   _Float16* __restrict__ Wh,
                                                   long n) {
  long i = ((long)blockIdx.x * 256 + threadIdx.x) * 4;
  if (i + 3 < n) {
    float4 f = *(const float4*)(W + i);
    v4h h;
    h.x = (_Float16)f.x; h.y = (_Float16)f.y;
    h.z = (_Float16)f.z; h.w = (_Float16)f.w;
    *(v4h*)(Wh + i) = h;
  }
}

// ---------------------------------------------------------------------------
// Block reduction helper (wave32): three simultaneous sums.
// ---------------------------------------------------------------------------
__device__ __forceinline__ void warp_red3(float& a, float& b, float& c) {
  #pragma unroll
  for (int o = 16; o > 0; o >>= 1) {
    a += __shfl_down(a, o, 32);
    b += __shfl_down(b, o, 32);
    c += __shfl_down(c, o, 32);
  }
}

// ---------------------------------------------------------------------------
// Kernel B: v = log0(x) rowwise, cast to f16.  One 256-thread block per row.
// ---------------------------------------------------------------------------
__global__ __launch_bounds__(256) void pvfc_log0(const float* __restrict__ x,
                                                 _Float16* __restrict__ vh,
                                                 int D) {
  const long row = blockIdx.x;
  const float* xr = x + row * (long)D;
  const int t = threadIdx.x;

  float xv[8];
  float ss = 0.f, dummy0 = 0.f, dummy1 = 0.f;
  #pragma unroll
  for (int i = 0; i < 2; ++i) {
    float4 f = *(const float4*)(xr + (t + i * 256) * 4);
    xv[i * 4 + 0] = f.x; xv[i * 4 + 1] = f.y;
    xv[i * 4 + 2] = f.z; xv[i * 4 + 3] = f.w;
    ss += f.x * f.x + f.y * f.y + f.z * f.z + f.w * f.w;
  }

  __shared__ float sm[8];
  __shared__ float scoef;
  warp_red3(ss, dummy0, dummy1);
  if ((t & 31) == 0) sm[t >> 5] = ss;
  __syncthreads();
  if (t == 0) {
    float tot = 0.f;
    #pragma unroll
    for (int w = 0; w < 8; ++w) tot += sm[w];
    float s = sqrtf(tot);                       // S == 1
    scoef = asinhf(s) / fmaxf(s, EPSF);
  }
  __syncthreads();
  const float coef = scoef;

  _Float16* vr = vh + row * (long)D;
  #pragma unroll
  for (int i = 0; i < 2; ++i) {
    v4h h;
    h.x = (_Float16)(coef * xv[i * 4 + 0]);
    h.y = (_Float16)(coef * xv[i * 4 + 1]);
    h.z = (_Float16)(coef * xv[i * 4 + 2]);
    h.w = (_Float16)(coef * xv[i * 4 + 3]);
    *(v4h*)(vr + (t + i * 256) * 4) = h;
  }
}

// ---------------------------------------------------------------------------
// Kernel C: u = v @ W^T via WMMA f16->f32, double-buffered async LDS tiles.
// Block: 256 threads = 8 wave32.  Tile: 128(M) x 128(N), K-step 32.
// LDS: 2 x (A[128][32] + B[128][32]) f16 = 32KB.
// Wave w computes rows [16w, 16w+16) x all 128 cols as 8 accumulators.
// ---------------------------------------------------------------------------
__global__ __launch_bounds__(256) void pvfc_gemm_wmma(
    const _Float16* __restrict__ Aglb,   // vh  [N, K] row-major
    const _Float16* __restrict__ Bglb,   // Wh  [D_OUT, K] row-major (B = W^T)
    float* __restrict__ Uglb,            // u   [N, D_OUT]
    int K, int ldu) {
  __shared__ _Float16 As[2][128 * 32];
  __shared__ _Float16 Bs[2][128 * 32];

  const int t    = threadIdx.x;
  const int lane = t & 31;
  const int wave = t >> 5;
  const int l15  = lane & 15;
  const int l4   = lane >> 4;

  const long m0 = (long)blockIdx.x * 128;
  const long n0 = (long)blockIdx.y * 128;

  const v8f vzero = {};
  v8f acc[8];
  #pragma unroll
  for (int j = 0; j < 8; ++j) acc[j] = vzero;

  // One K-step's WMMA block on LDS buffer `buf` (compile-time constant).
  // A fragment: 16x32 f16.  ISA layout: lanes 0-15 hold M=lane, K chunks
  // {0..7,16..23}; lanes 16-31 hold M=lane-16, K chunks {8..15,24..31}.
#define PVFC_COMPUTE(buf)                                                   \
  {                                                                         \
    union Frag { v16h v; v8h h[2]; };                                       \
    Frag a_;                                                                \
    const _Float16* arow_ = &As[buf][(wave * 16 + l15) * 32];               \
    a_.h[0] = *(const v8h*)(arow_ + l4 * 8);                                \
    a_.h[1] = *(const v8h*)(arow_ + 16 + l4 * 8);                           \
    _Pragma("unroll")                                                       \
    for (int j_ = 0; j_ < 8; ++j_) {                                        \
      Frag b_;                                                              \
      const _Float16* brow_ = &Bs[buf][(j_ * 16 + l15) * 32];               \
      b_.h[0] = *(const v8h*)(brow_ + l4 * 8);                              \
      b_.h[1] = *(const v8h*)(brow_ + 16 + l4 * 8);                         \
      acc[j_] = __builtin_amdgcn_wmma_f32_16x16x32_f16(                     \
          false, a_.v, false, b_.v, (short)0, acc[j_], false, false);       \
    }                                                                       \
  }

#ifdef PVFC_USE_ASYNC
  // Per tile: 512 x 16B chunks per matrix, 2 per thread per matrix (4 total).
#define PVFC_ISSUE_TILE(k0, buf)                                            \
  {                                                                         \
    _Pragma("unroll")                                                       \
    for (int i_ = 0; i_ < 2; ++i_) {                                        \
      const int q_ = t + i_ * 256;                                          \
      const int r_ = q_ >> 2;                                               \
      const int c_ = (q_ & 3) * 8;                                          \
      const _Float16* ga_ = Aglb + (m0 + r_) * (long)K + (k0) + c_;         \
      const _Float16* gb_ = Bglb + (n0 + r_) * (long)K + (k0) + c_;         \
      __builtin_amdgcn_global_load_async_to_lds_b128(                       \
          (pvfc_g128_t)ga_, (pvfc_l128_t)&As[buf][q_ * 8], 0, 0);           \
      __builtin_amdgcn_global_load_async_to_lds_b128(                       \
          (pvfc_g128_t)gb_, (pvfc_l128_t)&Bs[buf][q_ * 8], 0, 0);           \
    }                                                                       \
  }

  // Branch-free 2-stage pipeline; requires K % 64 == 0 (reference: K = 2048).
  PVFC_ISSUE_TILE(0, 0);
  int k0 = 0;
  for (; k0 + 64 < K; k0 += 64) {
    PVFC_ISSUE_TILE(k0 + 32, 1);   // next tile in flight
    pvfc_async_wait4();            // tile k0 landed
    __syncthreads();
    PVFC_COMPUTE(0);
    __syncthreads();               // buf0 free for refill

    PVFC_ISSUE_TILE(k0 + 64, 0);
    pvfc_async_wait4();            // tile k0+32 landed
    __syncthreads();
    PVFC_COMPUTE(1);
    __syncthreads();               // buf1 free for refill
  }
  // tail pair: tiles k0 (already issued into buf0) and k0+32
  PVFC_ISSUE_TILE(k0 + 32, 1);
  pvfc_async_wait4();
  __syncthreads();
  PVFC_COMPUTE(0);
  __syncthreads();
  pvfc_async_wait0();
  __syncthreads();
  PVFC_COMPUTE(1);
#undef PVFC_ISSUE_TILE
#else
  // ---- fallback: synchronous loads through VGPRs, single buffer ----
  for (int k0 = 0; k0 < K; k0 += 32) {
    __syncthreads();
    #pragma unroll
    for (int i = 0; i < 2; ++i) {
      const int q = t + i * 256;
      const int r = q >> 2;
      const int c = (q & 3) * 8;
      *(v8h*)&As[0][q * 8] = *(const v8h*)(Aglb + (m0 + r) * (long)K + k0 + c);
      *(v8h*)&Bs[0][q * 8] = *(const v8h*)(Bglb + (n0 + r) * (long)K + k0 + c);
    }
    __syncthreads();
    PVFC_COMPUTE(0);
  }
#endif
#undef PVFC_COMPUTE

  // C/D layout: VGPR r, lanes 0-15 -> M = r, lanes 16-31 -> M = r + 8; N = l15.
  const long mrow = m0 + wave * 16 + l4 * 8;
  #pragma unroll
  for (int j = 0; j < 8; ++j) {
    const long ncol = n0 + j * 16 + l15;
    #pragma unroll
    for (int r = 0; r < 8; ++r) {
      Uglb[(mrow + r) * (long)ldu + ncol] = acc[j][r];
    }
  }
}

// ---------------------------------------------------------------------------
// Kernel D: fused proj_tan0 + exp0 + pt_0_to_y + exp_map + gyro_add.
// Per row, everything reduces to out = P*u + Q*b with scalars from
// su2 = sum(u^2), sub = sum(u*b), sbb = sum(b^2).  In-place on d_out.
// ---------------------------------------------------------------------------
__global__ __launch_bounds__(256) void pvfc_epilogue(float* __restrict__ U,
                                                     const float* __restrict__ bias,
                                                     int D) {
  const long row = blockIdx.x;
  float* ur = U + row * (long)D;
  const int t = threadIdx.x;

  float uv[8], bv[8];
  float su2 = 0.f, sub = 0.f, sbb = 0.f;
  #pragma unroll
  for (int i = 0; i < 2; ++i) {
    const int idx = (t + i * 256) * 4;
    float4 fu = *(const float4*)(ur + idx);
    float4 fb = *(const float4*)(bias + idx);
    uv[i * 4 + 0] = fu.x; uv[i * 4 + 1] = fu.y; uv[i * 4 + 2] = fu.z; uv[i * 4 + 3] = fu.w;
    bv[i * 4 + 0] = fb.x; bv[i * 4 + 1] = fb.y; bv[i * 4 + 2] = fb.z; bv[i * 4 + 3] = fb.w;
    su2 += fu.x * fu.x + fu.y * fu.y + fu.z * fu.z + fu.w * fu.w;
    sub += fu.x * fb.x + fu.y * fb.y + fu.z * fb.z + fu.w * fb.w;
    sbb += fb.x * fb.x + fb.y * fb.y + fb.z * fb.z + fb.w * fb.w;
  }

  __shared__ float s0[8], s1[8], s2[8];
  __shared__ float sPQ[2];
  warp_red3(su2, sub, sbb);
  if ((t & 31) == 0) { s0[t >> 5] = su2; s1[t >> 5] = sub; s2[t >> 5] = sbb; }
  __syncthreads();
  if (t == 0) {
    float a0 = 0.f, a1 = 0.f, a2 = 0.f;
    #pragma unroll
    for (int w = 0; w < 8; ++w) { a0 += s0[w]; a1 += s1[w]; a2 += s2[w]; }
    // proj_tan0 + exp0 : y = cy * u
    const float r0  = sqrtf(a0);
    const float sc1 = fminf(20.f / fmaxf(r0, EPSF), 1.f);
    const float rp  = sc1 * r0;
    const float ce  = sinhf(rp) / fmaxf(rp, EPSF);
    const float cy  = ce * sc1;
    const float yy  = cy * cy * a0;       // sum(y^2)
    const float yb  = cy * a1;            // sum(y*b)
    const float sbbv = a2;                // sum(b^2)
    // pt_0_to_y : bt = b + aa*y
    const float bet = 1.f / fmaxf(sqrtf(1.f + yy), TINYF);
    const float cpt = bet / (1.f + bet);
    const float aa  = cpt * yb;
    // exp_map(y, bt)
    const float v2  = sbbv + 2.f * aa * yb + aa * aa * yy;
    const float dxv = yb + aa * yy;
    const float g   = fmaxf(v2 - dxv * dxv / (1.f + yy), TINYF);
    const float rg  = sqrtf(g);
    const float c1  = bet / (1.f + bet);
    const float t1  = fmaxf(1.f + bet, TINYF);
    const float c2  = -(bet * bet * bet) / (t1 * t1);
    const float lam = (1.f + bet) / fmaxf(bet, TINYF);
    const float z   = fminf(fmaxf(rg, -20.f), 20.f);
    const float shc = (fabsf(z) < EPSF) ? (1.f + z * z / 6.f)
                                        : (sinhf(z) / fmaxf(z, EPSF));
    const float L   = lam * shc;
    const float Wb  = L * c1;             // w = Wb*b + Wy*y
    const float Wy  = Wb * aa + L * c2 * dxv;
    // gyro_add(y, w)
    const float w2  = Wb * Wb * sbbv + 2.f * Wb * Wy * yb + Wy * Wy * yy;
    const float bw  = 1.f / fmaxf(sqrtf(1.f + w2), TINYF);
    const float xyv = Wb * yb + Wy * yy;
    const float cg  = bet / (1.f + bet) * xyv + (1.f - bw) / bw;
    sPQ[0] = (1.f + Wy + cg) * cy;        // P (multiplies u)
    sPQ[1] = Wb;                          // Q (multiplies b)
  }
  __syncthreads();
  const float P = sPQ[0], Q = sPQ[1];

  #pragma unroll
  for (int i = 0; i < 2; ++i) {
    const int idx = (t + i * 256) * 4;
    float4 o;
    o.x = P * uv[i * 4 + 0] + Q * bv[i * 4 + 0];
    o.y = P * uv[i * 4 + 1] + Q * bv[i * 4 + 1];
    o.z = P * uv[i * 4 + 2] + Q * bv[i * 4 + 2];
    o.w = P * uv[i * 4 + 3] + Q * bv[i * 4 + 3];
    *(float4*)(ur + idx) = o;
  }
}

// ---------------------------------------------------------------------------
// Host launcher
// ---------------------------------------------------------------------------
extern "C" void kernel_launch(void* const* d_in, const int* in_sizes, int n_in,
                              void* d_out, int out_size, void* d_ws, size_t ws_size,
                              hipStream_t stream) {
  const float* x = (const float*)d_in[0];
  const float* W = (const float*)d_in[1];
  const float* b = (const float*)d_in[2];
  float* out = (float*)d_out;

  const int D_OUT = in_sizes[2];                 // 2048
  const int D_IN  = in_sizes[1] / D_OUT;         // 2048
  const int N     = in_sizes[0] / D_IN;          // 16384

  // Workspace layout: Wh (f16 D_OUT*D_IN) | vh (f16 N*D_IN)  ~= 72 MB
  _Float16* Wh = (_Float16*)d_ws;
  _Float16* vh = Wh + (size_t)D_OUT * D_IN;

  const long nW = (long)D_OUT * D_IN;
  pvfc_cast_w<<<(unsigned)(nW / 1024), 256, 0, stream>>>(W, Wh, nW);
  pvfc_log0<<<N, 256, 0, stream>>>(x, vh, D_IN);

  dim3 grid(N / 128, D_OUT / 128);
  pvfc_gemm_wmma<<<grid, 256, 0, stream>>>(vh, Wh, out, D_IN, D_OUT);

  pvfc_epilogue<<<N, 256, 0, stream>>>(out, b, D_OUT);
}
